// MulMoE_7722351198588
// MI455X (gfx1250) — matrix operand order repaired and checked by reading, not compiled, not run on hardware
//
#include <hip/hip_runtime.h>
#include <hip/hip_bf16.h>

// ---------------- problem constants ----------------
#define BATCH   16384
#define NEXP    16
#define DIM     512
#define UNITS   256
#define FEAT    512
#define HEADS   4
#define KTOP    4
#define INV_T   100.0f      // 1/TEMP
#define BN_EPS  1e-5f

typedef __attribute__((ext_vector_type(16))) __bf16 v16bf;
typedef __attribute__((ext_vector_type(8)))  float  v8f;

union FragBF {
    uint4 q[2];   // 32 bytes
    v16bf v;
};

// exact single-instruction f32x2 -> packed bf16 (RNE); keeps SLP from
// re-pairing elements and emitting and/shl/or fixups
__device__ __forceinline__ unsigned pack_bf16(float lo, float hi) {
    unsigned r;
    asm("v_cvt_pk_bf16_f32 %0, %1, %2" : "=v"(r) : "v"(lo), "v"(hi));
    return r;
}

// LDS row geometry: 20 dwords (= 40 halves) per 32-half row -> 16B-aligned
// b128 fragment access, 20*col distinct mod 64 (conflict-free frag reads)
#define ROWU 20
#define MT   64     // M tile (A read exactly once from HBM)
#define EOP  260    // padded row stride for K2 expert tile (16B aligned, bank=4e+u)

// =====================================================================
// Kernel 0: W_exp [E,D,U] fp32 -> Wt [E,U,D] bf16 (packed pairs along k)
// =====================================================================
__global__ __launch_bounds__(256) void moe_convert_w(
    const float* __restrict__ W, unsigned* __restrict__ Wt)
{
    const size_t g = (size_t)blockIdx.x * 256 + threadIdx.x;  // 1M total
    const int e  = (int)(g >> 16);
    const int kp = (int)((g >> 8) & 255);
    const int u  = (int)(g & 255);
    const float a = W[((size_t)e * DIM + 2 * kp) * UNITS + u];
    const float b = W[((size_t)e * DIM + 2 * kp + 1) * UNITS + u];
    Wt[((size_t)e * UNITS + u) * (DIM / 2) + kp] = pack_bf16(a, b);
}

// =====================================================================
// Kernel 1: per-expert GEMM, tile 64(M) x 256(N), K chunks of 32,
//           v_wmma_f32_16x16x32_bf16, epilogue -> ws [B,E,U]
// =====================================================================
__global__ __launch_bounds__(256) void moe_expert_gemm(
    const float* __restrict__ inputs,    // [B, E*D] fp32
    const unsigned* __restrict__ Wt,     // [E, U, D/2] packed bf16
    const float* __restrict__ blin,
    const float* __restrict__ bnG,
    const float* __restrict__ bnB,
    const float* __restrict__ bnM,
    const float* __restrict__ bnV,
    const float* __restrict__ ebias,
    float* __restrict__ ws_eo)           // [B, E, U]
{
    __shared__ unsigned As[2][MT * ROWU];     // [m][k] bf16 pairs
    __shared__ unsigned Bs[2][UNITS * ROWU];  // [n][k] bf16 pairs

    const int e  = blockIdx.x >> 8;           // 16 experts
    const int m0 = (blockIdx.x & 255) << 6;   // 256 M-tiles of 64

    const int tid  = threadIdx.x;
    const int wave = tid >> 5;
    const int lane = tid & 31;
    const int l    = lane & 15;
    const int grp  = lane >> 4;
    const int wm   = wave >> 2;               // 2 M-groups of 2 subtiles
    const int wn   = wave & 3;                // 4 N-groups of 4 subtiles

    // staging roles (256 threads)
    const int arow = tid >> 2, aseg = tid & 3;   // A: 64 rows x (4x8 floats)

    const float*    aG = inputs + (size_t)(m0 + arow) * (NEXP * DIM) + (size_t)e * DIM + aseg * 8;
    const unsigned* bG = Wt + ((size_t)e * UNITS + tid) * (DIM / 2);   // one n-row per thread

    v8f acc[2][4];
#pragma unroll
    for (int i = 0; i < 2; ++i)
#pragma unroll
        for (int j = 0; j < 4; ++j)
#pragma unroll
            for (int r = 0; r < 8; ++r) acc[i][j][r] = 0.0f;

    // ---- prologue: stage chunk 0 ----
    {
        float4 a0 = *(const float4*)(aG);
        float4 a1 = *(const float4*)(aG + 4);
        uint4  bv[4];
#pragma unroll
        for (int w = 0; w < 4; ++w) bv[w] = *(const uint4*)(bG + w * 4);
        uint4 pa;
        pa.x = pack_bf16(a0.x, a0.y); pa.y = pack_bf16(a0.z, a0.w);
        pa.z = pack_bf16(a1.x, a1.y); pa.w = pack_bf16(a1.z, a1.w);
        *(uint4*)&As[0][arow * ROWU + aseg * 4] = pa;
#pragma unroll
        for (int w = 0; w < 4; ++w) *(uint4*)&Bs[0][tid * ROWU + w * 4] = bv[w];
    }

    // ---- main K loop: 16 chunks of 32 ----
    for (int kc = 0; kc < 16; ++kc) {
        __syncthreads();
        const int cur = kc & 1, nxt = cur ^ 1;
        const bool pre = (kc + 1) < 16;

        float4 a0, a1;
        uint4  bv[4];
        if (pre) {
            const int k1 = (kc + 1) * 32;
            a0 = *(const float4*)(aG + k1);
            a1 = *(const float4*)(aG + k1 + 4);
#pragma unroll
            for (int w = 0; w < 4; ++w) bv[w] = *(const uint4*)(bG + k1 / 2 + w * 4);
        }

        // load all 6 fragments, then 8 back-to-back WMMAs
        FragBF fa[2], fb[4];
#pragma unroll
        for (int i = 0; i < 2; ++i) {
            const unsigned* ab = &As[cur][(wm * 32 + i * 16 + l) * ROWU + grp * 4];
            fa[i].q[0] = *(const uint4*)(ab);
            fa[i].q[1] = *(const uint4*)(ab + 8);
        }
#pragma unroll
        for (int j = 0; j < 4; ++j) {
            const unsigned* bb = &Bs[cur][(wn * 64 + j * 16 + l) * ROWU + grp * 4];
            fb[j].q[0] = *(const uint4*)(bb);
            fb[j].q[1] = *(const uint4*)(bb + 8);
        }
#pragma unroll
        for (int i = 0; i < 2; ++i)
#pragma unroll
            for (int j = 0; j < 4; ++j)
                acc[i][j] = __builtin_amdgcn_wmma_f32_16x16x32_bf16(
                    false, fa[i].v, false, fb[j].v, (short)0, acc[i][j], false, false);

        if (pre) {
            uint4 pa;
            pa.x = pack_bf16(a0.x, a0.y); pa.y = pack_bf16(a0.z, a0.w);
            pa.z = pack_bf16(a1.x, a1.y); pa.w = pack_bf16(a1.z, a1.w);
            *(uint4*)&As[nxt][arow * ROWU + aseg * 4] = pa;
#pragma unroll
            for (int w = 0; w < 4; ++w) *(uint4*)&Bs[nxt][tid * ROWU + w * 4] = bv[w];
        }
    }

    // ---- epilogue: +b_lin -> BN(eval) -> ReLU -> +expert_bias -> LeakyReLU ----
#pragma unroll
    for (int j = 0; j < 4; ++j) {
        const int u  = wn * 64 + j * 16 + l;
        const int pu = e * UNITS + u;
        const float mean = bnM[pu];
        const float rstd = rsqrtf(bnV[pu] + BN_EPS);
        const float gsc  = rstd * bnG[pu];
        const float beta = bnB[pu];
        const float bl   = blin[pu];
        const float eb   = ebias[pu];
#pragma unroll
        for (int i = 0; i < 2; ++i) {
#pragma unroll
            for (int r = 0; r < 8; ++r) {
                const int row = m0 + wm * 32 + i * 16 + grp * 8 + r;
                float h = acc[i][j][r] + bl;
                h = (h - mean) * gsc + beta;
                h = fmaxf(h, 0.0f);
                h += eb;
                const float o = (h >= 0.0f) ? h : 0.01f * h;
                ws_eo[((size_t)row * NEXP + e) * UNITS + u] = o;
            }
        }
    }
}

// =====================================================================
// Kernel 2 (fused): gating -> top4 -> softmax -> combine -> out1,
//                   plus transpose [B,E,U]->[B,U,E] -> out2.
// =====================================================================
__global__ __launch_bounds__(256) void moe_gate_combine_t(
    const float* __restrict__ feat,
    const float* __restrict__ Wg,
    const float* __restrict__ bg,
    const float* __restrict__ gbias,
    const float* __restrict__ gwts,
    const float* __restrict__ ws_eo,     // [B, E, U]
    float* __restrict__ out1,            // [B, H*U]
    float* __restrict__ out2)            // [B, U, E]
{
    __shared__ float sf[4][FEAT];
    __shared__ float seo[4][NEXP * EOP];  // padded: bank = (4e+u) mod 64, 16B rows
    __shared__ float part[4][64][4];
    __shared__ float swg[4][64];
    __shared__ float sgw[NEXP];
    __shared__ float snv[16][KTOP];
    __shared__ int   sidx[16][KTOP];

    const int t  = threadIdx.x;
    const int b0 = blockIdx.x * 4;

    // stage features (b128)
#pragma unroll
    for (int i = t; i < (4 * FEAT) / 4; i += 256) {
        const int r4 = i * 4;
        const int bi = r4 >> 9, f = r4 & (FEAT - 1);
        *(float4*)&sf[bi][f] = *(const float4*)(feat + (size_t)(b0 + bi) * FEAT + f);
    }
    // stage the 4x[16][256] expert tile (b128 global + b128 LDS)
#pragma unroll
    for (int i = t; i < (4 * NEXP * UNITS) / 4; i += 256) {
        const int r4 = i * 4;
        const int bi = r4 >> 12, rem = r4 & 4095;
        const int e = rem >> 8, u = rem & 255;
        *(float4*)&seo[bi][e * EOP + u] =
            *(const float4*)(ws_eo + (size_t)(b0 + bi) * (NEXP * UNITS) + rem);
    }
    if (t == 0) {
        float m = -3.4e38f;
        for (int e2 = 0; e2 < NEXP; ++e2) m = fmaxf(m, gwts[e2]);
        float s = 0.0f, ex[NEXP];
        for (int e2 = 0; e2 < NEXP; ++e2) { ex[e2] = __expf((gwts[e2] - m) * INV_T); s += ex[e2]; }
        const float inv = 1.0f / s;
        for (int e2 = 0; e2 < NEXP; ++e2) sgw[e2] = ex[e2] * inv;
    }
    __syncthreads();

    // partial gate dot products: thread = (quarter q, h*16+e)
    {
        const int he = t & 63, q = t >> 6;
        const int h = he >> 4, e2 = he & 15;
        float a0 = 0, a1 = 0, a2 = 0, a3 = 0;
        const float* wcol = Wg + ((size_t)h * FEAT + q * 128) * NEXP + e2;
        const int fb = q * 128;
        for (int f = 0; f < 128; ++f) {
            const float w = wcol[f * NEXP];
            a0 += sf[0][fb + f] * w;
            a1 += sf[1][fb + f] * w;
            a2 += sf[2][fb + f] * w;
            a3 += sf[3][fb + f] * w;
        }
        part[q][he][0] = a0; part[q][he][1] = a1; part[q][he][2] = a2; part[q][he][3] = a3;
    }
    __syncthreads();

    {
        const int bi = t >> 6, he = t & 63;
        const int h = he >> 4, e2 = he & 15;
        float g = part[0][he][bi] + part[1][he][bi] + part[2][he][bi] + part[3][he][bi];
        g += bg[h * NEXP + e2] + gbias[h * NEXP + e2];
        g = fmaxf(g, 0.0f);
        swg[bi][he] = g * sgw[e2];
    }
    __syncthreads();

    // top-4 of 16 (ties -> lower index) + softmax(top/T)
    if (t < 16) {
        const int bi = t >> 2, h = t & 3;
        float v[NEXP];
        for (int e2 = 0; e2 < NEXP; ++e2) v[e2] = swg[bi][h * 16 + e2];
        float tv[KTOP]; int ti[KTOP];
        for (int k = 0; k < KTOP; ++k) {
            float best = -1.0f; int bix = 0;
            for (int e2 = 0; e2 < NEXP; ++e2)
                if (v[e2] > best) { best = v[e2]; bix = e2; }
            tv[k] = best; ti[k] = bix; v[bix] = -1.0f;
        }
        const float m = tv[0];
        float s = 0.0f, ex[KTOP];
        for (int k = 0; k < KTOP; ++k) { ex[k] = __expf((tv[k] - m) * INV_T); s += ex[k]; }
        const float inv = 1.0f / s;
        for (int k = 0; k < KTOP; ++k) { snv[t][k] = ex[k] * inv; sidx[t][k] = ti[k]; }
    }
    __syncthreads();

    // combine from LDS -> out1  (thread = u; conflict-free: bank = 4*idx + u)
    for (int p = 0; p < 16; ++p) {
        const int bi = p >> 2, h = p & 3;
        const size_t b = (size_t)(b0 + bi);
        float s = 0.0f;
#pragma unroll
        for (int k = 0; k < KTOP; ++k)
            s += snv[p][k] * seo[bi][sidx[p][k] * EOP + t];
        out1[b * (HEADS * UNITS) + h * UNITS + t] = s;
    }

    // transpose from LDS -> out2 (b128 coalesced global, conflict-free LDS)
#pragma unroll
    for (int i = t; i < (4 * NEXP * UNITS) / 4; i += 256) {
        const int r4 = i * 4;
        const int bi = r4 >> 12, rem = r4 & 4095;
        const int u = rem >> 4, e0 = rem & 15;        // e0 in {0,4,8,12}
        float4 v;
        v.x = seo[bi][(e0 + 0) * EOP + u];
        v.y = seo[bi][(e0 + 1) * EOP + u];
        v.z = seo[bi][(e0 + 2) * EOP + u];
        v.w = seo[bi][(e0 + 3) * EOP + u];
        *(float4*)(out2 + (size_t)(b0 + bi) * (NEXP * UNITS) + rem) = v;
    }
}

// =====================================================================
extern "C" void kernel_launch(void* const* d_in, const int* in_sizes, int n_in,
                              void* d_out, int out_size, void* d_ws, size_t ws_size,
                              hipStream_t stream) {
    const float* feature   = (const float*)d_in[0];
    const float* inputs    = (const float*)d_in[1];
    const float* W_exp     = (const float*)d_in[2];
    const float* b_lin     = (const float*)d_in[3];
    const float* bn_gamma  = (const float*)d_in[4];
    const float* bn_beta   = (const float*)d_in[5];
    const float* bn_mean   = (const float*)d_in[6];
    const float* bn_var    = (const float*)d_in[7];
    const float* exp_bias  = (const float*)d_in[8];
    const float* W_gate    = (const float*)d_in[9];
    const float* b_gate    = (const float*)d_in[10];
    const float* gate_bias = (const float*)d_in[11];
    const float* gweights  = (const float*)d_in[12];

    float* out1 = (float*)d_out;                                   // [B, H*U]
    float* out2 = (float*)d_out + (size_t)BATCH * HEADS * UNITS;   // [B, U, E]

    float*    ws_eo = (float*)d_ws;                                // [B, E, U] fp32
    unsigned* wt    = (unsigned*)((char*)d_ws +
                       (size_t)BATCH * NEXP * UNITS * sizeof(float)); // [E,U,D/2]

    // K0: convert + transpose expert weights to bf16 [E,U,D]
    moe_convert_w<<<(NEXP * UNITS * (DIM / 2)) / 256, 256, 0, stream>>>(W_exp, wt);

    // K1: expert GEMMs (E * B/64 = 4096 workgroups); A read exactly once
    moe_expert_gemm<<<NEXP * (BATCH / MT), 256, 0, stream>>>(
        inputs, wt, b_lin, bn_gamma, bn_beta, bn_mean, bn_var, exp_bias, ws_eo);

    // K2: fused gating + topk + combine + transpose
    moe_gate_combine_t<<<BATCH / 4, 256, 0, stream>>>(
        feature, W_gate, b_gate, gate_bias, gweights, ws_eo, out1, out2);
}
